// Attention_83640193123012
// MI455X (gfx1250) — matrix operand order repaired and checked by reading, not compiled, not run on hardware
//
#include <hip/hip_runtime.h>
#include <hip/hip_bf16.h>

// ---------------------------------------------------------------------------
// Types for CDNA5 WMMA (wave32, 16x16x32 bf16 -> f32)
// ---------------------------------------------------------------------------
typedef __attribute__((ext_vector_type(16))) __bf16 v16bf;
typedef __attribute__((ext_vector_type(8)))  float  v8f;
typedef __attribute__((ext_vector_type(4)))  int    v4i;

union FragU {
    unsigned int u[8];
    v16bf v;
};

__device__ __forceinline__ v8f wmma_bf16(v16bf a, v16bf b, v8f c) {
    return __builtin_amdgcn_wmma_f32_16x16x32_bf16(
        /*neg_a=*/false, a, /*neg_b=*/false, b,
        /*c_mod=*/(short)0, c, /*reuse_a=*/false, /*reuse_b=*/false);
}

__device__ __forceinline__ unsigned short f2bf(float x) {
    unsigned int u = __float_as_uint(x);
    unsigned int r = u + 0x7FFFu + ((u >> 16) & 1u);   // round-to-nearest-even
    return (unsigned short)(r >> 16);
}

// ---------------------------------------------------------------------------
// Async global -> LDS (CDNA5 GLOBAL_LOAD_ASYNC_TO_LDS_B128, ASYNCcnt-tracked)
// Builtin exists (probe-confirmed via __has_builtin in the previous round);
// expected proto: (v4i addrspace(1)*, v4i addrspace(3)*, imm offset, imm cpol)
// ---------------------------------------------------------------------------
#if defined(__has_builtin)
# if __has_builtin(__builtin_amdgcn_global_load_async_to_lds_b128)
#  define HAVE_ASYNC_LDS 1
# endif
#endif

#ifdef HAVE_ASYNC_LDS
__device__ __forceinline__ void async_copy16(const void* g, void* l) {
    __builtin_amdgcn_global_load_async_to_lds_b128(
        (__attribute__((address_space(1))) v4i*)g,
        (__attribute__((address_space(3))) v4i*)l,
        /*offset=*/0, /*cpol=*/0);
}
__device__ __forceinline__ void async_wait0() {
# if __has_builtin(__builtin_amdgcn_s_wait_asynccnt)
    __builtin_amdgcn_s_wait_asynccnt(0);
# else
    asm volatile("s_wait_asynccnt 0x0" ::: "memory");
# endif
}
#endif

// A fragment (16x32 bf16, M x K). lane&15 = row M; lanes 0-15 hold K
// {0..7,16..23}, lanes 16-31 hold K {8..15,24..31}. lds = u32 view, row-major.
__device__ __forceinline__ v16bf load_frag_a(const unsigned int* lds, int row0,
                                             int strideU, int k0, int lane) {
    FragU f;
    int m  = lane & 15;
    int hi = lane >> 4;
#pragma unroll
    for (int v = 0; v < 8; ++v) {
        int kk = k0 + ((v < 4) ? 0 : 16) + hi * 8 + 2 * (v & 3);
        f.u[v] = lds[(row0 + m) * strideU + (kk >> 1)];
    }
    return f.v;
}

// B fragment (32x16 bf16, K x N) where B[k][n] = T[n][k] for row-major tile T.
// lane&15 = col N; lanes 0-15 hold K {0..7,16..23}, lanes 16-31 hold K {8..15,24..31}.
__device__ __forceinline__ v16bf load_frag_b(const unsigned int* lds, int n0,
                                             int strideU, int k0, int lane) {
    FragU f;
    int n  = lane & 15;
    int hi = lane >> 4;
#pragma unroll
    for (int v = 0; v < 8; ++v) {
        int kk = k0 + ((v < 4) ? 0 : 16) + hi * 8 + 2 * (v & 3);
        f.u[v] = lds[(n0 + n) * strideU + (kk >> 1)];
    }
    return f.v;
}

// ---------------------------------------------------------------------------
// Kernel 1: fp32 -> bf16 convert
// ---------------------------------------------------------------------------
__global__ __launch_bounds__(256) void cvt_f32_bf16(const float* __restrict__ src,
                                                    unsigned short* __restrict__ dst,
                                                    int n) {
    int i = blockIdx.x * 256 + threadIdx.x;
    int stride = gridDim.x * 256;
    for (; i < n; i += stride) dst[i] = f2bf(src[i]);
}

// ---------------------------------------------------------------------------
// Kernel 2/5: GEMM  C[M x N] = A[M x K](bf16) * W[N x K]^T(bf16), fp32 out.
// 128 threads (4 waves), 128x64 block tile, wave -> 32x64, K-chunks of 32.
// Async path: double-buffered LDS with GLOBAL_LOAD_ASYNC_TO_LDS_B128.
// ---------------------------------------------------------------------------
__global__ __launch_bounds__(128) void gemm_bf16_wmma(
        const unsigned short* __restrict__ A, int lda,
        const unsigned short* __restrict__ W, int ldw,
        float* __restrict__ C, int ldc, int K) {
#ifdef HAVE_ASYNC_LDS
    __shared__ unsigned int As[2][128 * 20];   // 128 rows x (32 bf16 + pad) per buf
    __shared__ unsigned int Ws[2][64 * 20];
#else
    __shared__ unsigned int As[1][128 * 20];
    __shared__ unsigned int Ws[1][64 * 20];
#endif
    const int tid  = threadIdx.x;
    const int lane = tid & 31;
    const int w    = tid >> 5;
    const int m0   = blockIdx.y * 128;
    const int n0g  = blockIdx.x * 64;

    const unsigned int* Au = (const unsigned int*)A;
    const unsigned int* Wu = (const unsigned int*)W;
    const int ldau = lda >> 1;
    const int ldwu = ldw >> 1;

    v8f acc[2][4] = {};

#ifdef HAVE_ASYNC_LDS
    // Prologue: async-load tile 0 into buffer 0.
#pragma unroll
    for (int i = 0; i < 4; ++i) {                    // A: 512 b128 chunks
        int c = tid + i * 128;
        int r = c >> 2, q4 = (c & 3) * 4;
        async_copy16(&Au[(m0 + r) * ldau + q4], &As[0][r * 20 + q4]);
    }
#pragma unroll
    for (int i = 0; i < 2; ++i) {                    // W: 256 b128 chunks
        int c = tid + i * 128;
        int r = c >> 2, q4 = (c & 3) * 4;
        async_copy16(&Wu[(n0g + r) * ldwu + q4], &Ws[0][r * 20 + q4]);
    }
    async_wait0();
    __syncthreads();

    const int nT = K >> 5;
    for (int t = 0; t < nT; ++t) {
        const int cur = t & 1;
        const int nxt = cur ^ 1;
        if (t + 1 < nT) {                            // prefetch next K-chunk
            int kh = ((t + 1) << 5) >> 1;            // u32 offset
#pragma unroll
            for (int i = 0; i < 4; ++i) {
                int c = tid + i * 128;
                int r = c >> 2, q4 = (c & 3) * 4;
                async_copy16(&Au[(m0 + r) * ldau + kh + q4],
                             &As[nxt][r * 20 + q4]);
            }
#pragma unroll
            for (int i = 0; i < 2; ++i) {
                int c = tid + i * 128;
                int r = c >> 2, q4 = (c & 3) * 4;
                async_copy16(&Wu[(n0g + r) * ldwu + kh + q4],
                             &Ws[nxt][r * 20 + q4]);
            }
        }

        v16bf a0 = load_frag_a(&As[cur][0], w * 32,      20, 0, lane);
        v16bf a1 = load_frag_a(&As[cur][0], w * 32 + 16, 20, 0, lane);
        v16bf b0 = load_frag_b(&Ws[cur][0],  0, 20, 0, lane);
        v16bf b1 = load_frag_b(&Ws[cur][0], 16, 20, 0, lane);
        v16bf b2 = load_frag_b(&Ws[cur][0], 32, 20, 0, lane);
        v16bf b3 = load_frag_b(&Ws[cur][0], 48, 20, 0, lane);
        acc[0][0] = wmma_bf16(a0, b0, acc[0][0]);
        acc[1][0] = wmma_bf16(a1, b0, acc[1][0]);
        acc[0][1] = wmma_bf16(a0, b1, acc[0][1]);
        acc[1][1] = wmma_bf16(a1, b1, acc[1][1]);
        acc[0][2] = wmma_bf16(a0, b2, acc[0][2]);
        acc[1][2] = wmma_bf16(a1, b2, acc[1][2]);
        acc[0][3] = wmma_bf16(a0, b3, acc[0][3]);
        acc[1][3] = wmma_bf16(a1, b3, acc[1][3]);

        async_wait0();          // next buffer landed (overlapped with WMMAs)
        __syncthreads();        // everyone done reading cur, next visible
    }
#else
    for (int k0 = 0; k0 < K; k0 += 32) {
        __syncthreads();
#pragma unroll
        for (int i = 0; i < 16; ++i) {               // A: 128 rows x 16 u32
            int idx = tid + i * 128;
            int r = idx >> 4, cp = idx & 15;
            As[0][r * 20 + cp] = Au[(m0 + r) * ldau + (k0 >> 1) + cp];
        }
#pragma unroll
        for (int i = 0; i < 8; ++i) {                // W: 64 rows x 16 u32
            int idx = tid + i * 128;
            int r = idx >> 4, cp = idx & 15;
            Ws[0][r * 20 + cp] = Wu[(n0g + r) * ldwu + (k0 >> 1) + cp];
        }
        __syncthreads();

        v16bf a0 = load_frag_a(&As[0][0], w * 32,      20, 0, lane);
        v16bf a1 = load_frag_a(&As[0][0], w * 32 + 16, 20, 0, lane);
        v16bf b0 = load_frag_b(&Ws[0][0],  0, 20, 0, lane);
        v16bf b1 = load_frag_b(&Ws[0][0], 16, 20, 0, lane);
        v16bf b2 = load_frag_b(&Ws[0][0], 32, 20, 0, lane);
        v16bf b3 = load_frag_b(&Ws[0][0], 48, 20, 0, lane);
        acc[0][0] = wmma_bf16(a0, b0, acc[0][0]);
        acc[1][0] = wmma_bf16(a1, b0, acc[1][0]);
        acc[0][1] = wmma_bf16(a0, b1, acc[0][1]);
        acc[1][1] = wmma_bf16(a1, b1, acc[1][1]);
        acc[0][2] = wmma_bf16(a0, b2, acc[0][2]);
        acc[1][2] = wmma_bf16(a1, b2, acc[1][2]);
        acc[0][3] = wmma_bf16(a0, b3, acc[0][3]);
        acc[1][3] = wmma_bf16(a1, b3, acc[1][3]);
    }
#endif

    const int n  = lane & 15;
    const int hi = lane >> 4;
#pragma unroll
    for (int r = 0; r < 2; ++r) {
#pragma unroll
        for (int v = 0; v < 8; ++v) {
            int row = m0 + w * 32 + r * 16 + hi * 8 + v;
            float* Cr = C + (long)row * ldc + n0g + n;
            Cr[0]  = acc[r][0][v];
            Cr[16] = acc[r][1][v];
            Cr[32] = acc[r][2][v];
            Cr[48] = acc[r][3][v];
        }
    }
}

// ---------------------------------------------------------------------------
// Kernel 3: RoPE on Q (cols 0..1023) and K (1024..1279), plain convert of V,
// fp32 staging -> bf16. One thread per element pair. Rows = 4096, cols = 1536.
// ---------------------------------------------------------------------------
__global__ __launch_bounds__(256) void rope_cvt(
        const float* __restrict__ qkv,
        const float* __restrict__ fcos,
        const float* __restrict__ fsin,
        unsigned short* __restrict__ out) {
    int p   = blockIdx.x * 256 + threadIdx.x;   // pair index, 4096*768 total
    int row = p / 768;
    int c2  = (p % 768) * 2;
    int s   = row & 2047;

    const float* src = qkv + (long)row * 1536 + c2;
    unsigned short* dst = out + (long)row * 1536 + c2;
    float xr = src[0], xi = src[1];

    if (c2 < 1280) {                     // Q or K region: apply RoPE
        int i  = (c2 & 63) >> 1;
        float c  = fcos[s * 32 + i];
        float sn = fsin[s * 32 + i];
        float orr = xr * c - xi * sn;
        float oii = xr * sn + xi * c;
        dst[0] = f2bf(orr);
        dst[1] = f2bf(oii);
    } else {                             // V region: straight convert
        dst[0] = f2bf(xr);
        dst[1] = f2bf(xi);
    }
}

// ---------------------------------------------------------------------------
// Kernel 4: causal flash attention (GQA). qkv bf16 layout: [4096][1536],
// Q at col h*64, K at 1024+kv*64, V at 1280+kv*64.  Output bf16 [4096][1024].
// Grid: B*H*(S/64) = 1024 WGs of 128 threads; wave -> 16 q rows.
// ---------------------------------------------------------------------------
__global__ __launch_bounds__(128) void attn_flash(
        const unsigned short* __restrict__ qkv,
        unsigned short* __restrict__ outb) {
    __shared__ unsigned int  Qs[64 * 36];       // 64 x 72 bf16 (64 used + 8 pad)
    __shared__ unsigned int  Ks[64 * 36];
    __shared__ unsigned short Vts[64 * 72];     // transposed: [d][s]
    __shared__ unsigned short Ps[4][16 * 72];   // per-wave P strip

    const int tid  = threadIdx.x;
    const int lane = tid & 31;
    const int w    = tid >> 5;
    const int qt   = blockIdx.x & 31;
    const int h    = (blockIdx.x >> 5) & 15;
    const int bidx = blockIdx.x >> 9;
    const int kv   = h >> 2;                    // H=16, NREP=4

    const unsigned short* Vg = qkv + (long)bidx * 2048 * 1536 + 1280 + kv * 64;
    const unsigned int* Qgu =
        (const unsigned int*)(qkv + (long)bidx * 2048 * 1536 + h * 64);
    const unsigned int* Kgu =
        (const unsigned int*)(qkv + (long)bidx * 2048 * 1536 + 1024 + kv * 64);

    // Load Q tile (64 x 64 bf16 = 2048 u32 = 512 b128)
#ifdef HAVE_ASYNC_LDS
#pragma unroll
    for (int i = 0; i < 4; ++i) {
        int c = tid + i * 128;
        int r = c >> 3, q4 = (c & 7) * 4;
        async_copy16(&Qgu[(qt * 64 + r) * 768 + q4], &Qs[r * 36 + q4]);
    }
    async_wait0();
#else
#pragma unroll
    for (int i = 0; i < 16; ++i) {
        int idx  = tid + i * 128;
        int r    = idx >> 5;
        int cpos = idx & 31;
        Qs[r * 36 + cpos] = Qgu[(qt * 64 + r) * 768 + cpos];
    }
#endif
    __syncthreads();

    v16bf aq0 = load_frag_a(Qs, w * 16, 36, 0, lane);
    v16bf aq1 = load_frag_a(Qs, w * 16, 36, 32, lane);

    const int n  = lane & 15;
    const int hi = lane >> 4;
    const int qbase = qt * 64 + w * 16 + hi * 8;

    float mi[8], li[8];
#pragma unroll
    for (int v = 0; v < 8; ++v) { mi[v] = -1e30f; li[v] = 0.0f; }
    v8f o0 = {}, o1 = {}, o2 = {}, o3 = {};

    for (int j = 0; j <= qt; ++j) {
        __syncthreads();
        // K tile row-major (s, hh)
#ifdef HAVE_ASYNC_LDS
#pragma unroll
        for (int i = 0; i < 4; ++i) {
            int c = tid + i * 128;
            int r = c >> 3, q4 = (c & 7) * 4;
            async_copy16(&Kgu[(j * 64 + r) * 768 + q4], &Ks[r * 36 + q4]);
        }
#else
#pragma unroll
        for (int i = 0; i < 16; ++i) {
            int idx  = tid + i * 128;
            int r    = idx >> 5;
            int cpos = idx & 31;
            Ks[r * 36 + cpos] = Kgu[(j * 64 + r) * 768 + cpos];
        }
#endif
        // V tile transposed into (d, s)
#pragma unroll
        for (int i = 0; i < 32; ++i) {
            int idx = tid + i * 128;
            int s   = idx >> 6;
            int d   = idx & 63;
            Vts[d * 72 + s] = Vg[(long)(j * 64 + s) * 1536 + d];
        }
#ifdef HAVE_ASYNC_LDS
        async_wait0();
#endif
        __syncthreads();

        // Scores S = Q * K^T  (16 x 64 per wave)
        v8f s0 = {}, s1 = {}, s2 = {}, s3 = {};
        {
            v16bf b0 = load_frag_b(Ks,  0, 36, 0, lane);
            v16bf b1 = load_frag_b(Ks, 16, 36, 0, lane);
            v16bf b2 = load_frag_b(Ks, 32, 36, 0, lane);
            v16bf b3 = load_frag_b(Ks, 48, 36, 0, lane);
            s0 = wmma_bf16(aq0, b0, s0);
            s1 = wmma_bf16(aq0, b1, s1);
            s2 = wmma_bf16(aq0, b2, s2);
            s3 = wmma_bf16(aq0, b3, s3);
            b0 = load_frag_b(Ks,  0, 36, 32, lane);
            b1 = load_frag_b(Ks, 16, 36, 32, lane);
            b2 = load_frag_b(Ks, 32, 36, 32, lane);
            b3 = load_frag_b(Ks, 48, 36, 32, lane);
            s0 = wmma_bf16(aq1, b0, s0);
            s1 = wmma_bf16(aq1, b1, s1);
            s2 = wmma_bf16(aq1, b2, s2);
            s3 = wmma_bf16(aq1, b3, s3);
        }

        // Online softmax (per-row stats across the 16-lane half owning the row)
        const int sj0 = j * 64 + n;
#pragma unroll
        for (int v = 0; v < 8; ++v) {
            int qrow = qbase + v;
            float x0 = s0[v] * 0.125f; if (sj0      > qrow) x0 = -1e30f;
            float x1 = s1[v] * 0.125f; if (sj0 + 16 > qrow) x1 = -1e30f;
            float x2 = s2[v] * 0.125f; if (sj0 + 32 > qrow) x2 = -1e30f;
            float x3 = s3[v] * 0.125f; if (sj0 + 48 > qrow) x3 = -1e30f;

            float rm = fmaxf(fmaxf(x0, x1), fmaxf(x2, x3));
            rm = fmaxf(rm, __shfl_xor(rm, 1, 32));
            rm = fmaxf(rm, __shfl_xor(rm, 2, 32));
            rm = fmaxf(rm, __shfl_xor(rm, 4, 32));
            rm = fmaxf(rm, __shfl_xor(rm, 8, 32));

            float mnew  = fmaxf(mi[v], rm);
            float alpha = __expf(mi[v] - mnew);
            float p0 = __expf(x0 - mnew);
            float p1 = __expf(x1 - mnew);
            float p2 = __expf(x2 - mnew);
            float p3 = __expf(x3 - mnew);
            float rs = p0 + p1 + p2 + p3;
            rs += __shfl_xor(rs, 1, 32);
            rs += __shfl_xor(rs, 2, 32);
            rs += __shfl_xor(rs, 4, 32);
            rs += __shfl_xor(rs, 8, 32);

            li[v] = li[v] * alpha + rs;
            mi[v] = mnew;
            o0[v] *= alpha; o1[v] *= alpha; o2[v] *= alpha; o3[v] *= alpha;

            int m = hi * 8 + v;
            unsigned short* pr = &Ps[w][m * 72];
            pr[n]      = f2bf(p0);
            pr[16 + n] = f2bf(p1);
            pr[32 + n] = f2bf(p2);
            pr[48 + n] = f2bf(p3);
        }

        // O += P * V   (reload P from this wave's LDS strip as A fragments)
        const unsigned int* Pu = (const unsigned int*)&Ps[w][0];
        const unsigned int* Vu = (const unsigned int*)Vts;
        v16bf ap0 = load_frag_a(Pu, 0, 36, 0, lane);
        v16bf ap1 = load_frag_a(Pu, 0, 36, 32, lane);
        {
            v16bf b0 = load_frag_b(Vu,  0, 36, 0, lane);
            v16bf b1 = load_frag_b(Vu, 16, 36, 0, lane);
            v16bf b2 = load_frag_b(Vu, 32, 36, 0, lane);
            v16bf b3 = load_frag_b(Vu, 48, 36, 0, lane);
            o0 = wmma_bf16(ap0, b0, o0);
            o1 = wmma_bf16(ap0, b1, o1);
            o2 = wmma_bf16(ap0, b2, o2);
            o3 = wmma_bf16(ap0, b3, o3);
            b0 = load_frag_b(Vu,  0, 36, 32, lane);
            b1 = load_frag_b(Vu, 16, 36, 32, lane);
            b2 = load_frag_b(Vu, 32, 36, 32, lane);
            b3 = load_frag_b(Vu, 48, 36, 32, lane);
            o0 = wmma_bf16(ap1, b0, o0);
            o1 = wmma_bf16(ap1, b1, o1);
            o2 = wmma_bf16(ap1, b2, o2);
            o3 = wmma_bf16(ap1, b3, o3);
        }
    }

    // Epilogue: O / l, store bf16 to attention buffer [4096][1024]
#pragma unroll
    for (int v = 0; v < 8; ++v) {
        float inv = 1.0f / li[v];
        long row = (long)bidx * 2048 + qt * 64 + w * 16 + hi * 8 + v;
        unsigned short* dst = outb + row * 1024 + h * 64 + n;
        dst[0]  = f2bf(o0[v] * inv);
        dst[16] = f2bf(o1[v] * inv);
        dst[32] = f2bf(o2[v] * inv);
        dst[48] = f2bf(o3[v] * inv);
    }
}

// ---------------------------------------------------------------------------
// Host-side orchestration
// ---------------------------------------------------------------------------
extern "C" void kernel_launch(void* const* d_in, const int* in_sizes, int n_in,
                              void* d_out, int out_size, void* d_ws, size_t ws_size,
                              hipStream_t stream) {
    const float* x  = (const float*)d_in[0];   // [2,2048,1024]
    const float* wq = (const float*)d_in[1];   // [1024,1024]
    const float* wk = (const float*)d_in[2];   // [256,1024]
    const float* wv = (const float*)d_in[3];   // [256,1024]
    const float* wo = (const float*)d_in[4];   // [1024,1024]
    const float* fc = (const float*)d_in[5];   // [2048,32]
    const float* fs = (const float*)d_in[6];   // [2048,32]

    char* ws = (char*)d_ws;
    // Workspace layout (bytes)
    unsigned short* xb   = (unsigned short*)(ws);                    //  8,388,608
    unsigned short* Wall = (unsigned short*)(ws + 8388608);          //  3,145,728 [wq;wk;wv]
    unsigned short* wob  = (unsigned short*)(ws + 11534336);         //  2,097,152
    float*          qkvf = (float*)        (ws + 13631488);          // 25,165,824
    unsigned short* qkvb = (unsigned short*)(ws + 38797312);         // 12,582,912
    unsigned short* attb = (unsigned short*)(ws + 51380224);         //  8,388,608

    // 1) bf16 conversions
    {
        int nx = 2 * 2048 * 1024;
        cvt_f32_bf16<<<(nx + 255) / 256, 256, 0, stream>>>(x, xb, nx);
        cvt_f32_bf16<<<(1024 * 1024 + 255) / 256, 256, 0, stream>>>(wq, Wall, 1024 * 1024);
        cvt_f32_bf16<<<(256 * 1024 + 255) / 256, 256, 0, stream>>>(wk, Wall + 1024 * 1024, 256 * 1024);
        cvt_f32_bf16<<<(256 * 1024 + 255) / 256, 256, 0, stream>>>(wv, Wall + 1280 * 1024, 256 * 1024);
        cvt_f32_bf16<<<(1024 * 1024 + 255) / 256, 256, 0, stream>>>(wo, wob, 1024 * 1024);
    }

    // 2) Fused QKV projection: [4096,1536] = xb[4096,1024] * Wall[1536,1024]^T
    gemm_bf16_wmma<<<dim3(1536 / 64, 4096 / 128), 128, 0, stream>>>(
        xb, 1024, Wall, 1024, qkvf, 1536, 1024);

    // 3) RoPE + bf16 pack
    rope_cvt<<<(4096 * 768) / 256, 256, 0, stream>>>(qkvf, fc, fs, qkvb);

    // 4) Causal flash attention (GQA), bf16 out
    attn_flash<<<2 * 16 * (2048 / 64), 128, 0, stream>>>(qkvb, attb);

    // 5) Output projection: d_out[4096,1024] = attb * wob^T
    gemm_bf16_wmma<<<dim3(1024 / 64, 4096 / 128), 128, 0, stream>>>(
        attb, 1024, wob, 1024, (float*)d_out, 1024, 1024);
}
